// AttentionBlock_11605001633909
// MI455X (gfx1250) — matrix operand order repaired and checked by reading, not compiled, not run on hardware
//
#include <hip/hip_runtime.h>
#include <hip/hip_bf16.h>

// Problem constants (match reference)
#define CDIM   256
#define NPOS   4096     // 64*64
#define BATCH  4
#define NGROUP 32
#define CPG    8        // CDIM / NGROUP

// padded LDS row strides (in halves)
#define QS_STR 264      // 528 B/row: 16B aligned, 132-dword stride (conflict-free)
#define KS_STR 264
#define VS_STR 40       // 80 B/row
#define PS_STR 40
#define OS_STR 40

typedef __attribute__((ext_vector_type(16))) _Float16 v16h;
typedef __attribute__((ext_vector_type(8)))  float    v8f;

union AF { v16h v; unsigned int u[8]; _Float16 h[16]; };

__device__ __forceinline__ float red16_max(float x) {
#pragma unroll
  for (int m = 1; m < 16; m <<= 1) x = fmaxf(x, __shfl_xor(x, m, 16));
  return x;
}
__device__ __forceinline__ float red16_sum(float x) {
#pragma unroll
  for (int m = 1; m < 16; m <<= 1) x += __shfl_xor(x, m, 16);
  return x;
}

// --- CDNA5 async LDS-DMA helpers (ASYNCcnt-tracked) -----------------------
__device__ __forceinline__ void async_b128(unsigned lds_addr, const void* gaddr) {
  asm volatile("global_load_async_to_lds_b128 %0, %1, off"
               :: "v"(lds_addr), "v"(gaddr) : "memory");
}
__device__ __forceinline__ void wait_async0()  { asm volatile("s_wait_asynccnt 0x0"  ::: "memory"); }
__device__ __forceinline__ void wait_async16() { asm volatile("s_wait_asynccnt 0x10" ::: "memory"); }
__device__ __forceinline__ unsigned lds_off(const void* p) {
  return (unsigned)(unsigned long long)p;   // low 32 bits of generic LDS ptr = LDS address
}

// ---------------------------------------------------------------------------
// Kernel 0: convert weights to f16
// ---------------------------------------------------------------------------
__global__ void cvt_weights(const float* __restrict__ wq, const float* __restrict__ wp,
                            _Float16* __restrict__ wq16, _Float16* __restrict__ wp16) {
  int i = blockIdx.x * 256 + threadIdx.x;
  if (i < 3 * CDIM * CDIM) wq16[i] = (_Float16)wq[i];
  if (i < CDIM * CDIM)     wp16[i] = (_Float16)wp[i];
}

// ---------------------------------------------------------------------------
// Kernel 1: GroupNorm statistics (mu, rstd) per (b, group)
// ---------------------------------------------------------------------------
__global__ void gn_stats(const float* __restrict__ x, float* __restrict__ stats) {
  int bg = blockIdx.x;                 // b*32 + g
  int b = bg >> 5, g = bg & 31;
  const float* p = x + ((size_t)b * CDIM + g * CPG) * NPOS;
  float s = 0.f, s2 = 0.f;
  for (int i = threadIdx.x; i < CPG * NPOS; i += blockDim.x) {
    float v = p[i]; s += v; s2 += v * v;
  }
#pragma unroll
  for (int m = 1; m < 32; m <<= 1) { s += __shfl_xor(s, m, 32); s2 += __shfl_xor(s2, m, 32); }
  __shared__ float rs[8], rs2[8];
  int wv = threadIdx.x >> 5, ln = threadIdx.x & 31;
  if (ln == 0) { rs[wv] = s; rs2[wv] = s2; }
  __syncthreads();
  if (threadIdx.x == 0) {
    float a = 0.f, a2 = 0.f;
    for (int i = 0; i < 8; i++) { a += rs[i]; a2 += rs2[i]; }
    float inv = 1.0f / (float)(CPG * NPOS);
    float mu = a * inv;
    float var = a2 * inv - mu * mu;
    stats[bg * 2 + 0] = mu;
    stats[bg * 2 + 1] = rsqrtf(var + 1e-6f);
  }
}

// ---------------------------------------------------------------------------
// Kernel 2: fused GroupNorm-apply + QKV GEMM (WMMA f16 -> f32)
// q,k written (B,N,C); v written (B,C,N) for conflict-free P*V operands.
// ---------------------------------------------------------------------------
__global__ __launch_bounds__(128) void qkv_gemm(
    const float* __restrict__ x, const float* __restrict__ gamma,
    const float* __restrict__ beta, const float* __restrict__ stats,
    const _Float16* __restrict__ w16, const float* __restrict__ bq,
    _Float16* __restrict__ qt, _Float16* __restrict__ kt, _Float16* __restrict__ vt) {
  const int b = blockIdx.z;
  const int obase = blockIdx.y * 64;
  const int nbase = blockIdx.x * 64;
  const int t = threadIdx.x;
  const int wv = t >> 5, ln = t & 31;
  const int col = ln & 15, half = ln >> 4;

  __shared__ _Float16 hs[64][34];    // [n_local][c_local], padded for banks

  v8f zero = {};
  v8f acc[4];
#pragma unroll
  for (int i = 0; i < 4; i++) acc[i] = zero;

  for (int cc = 0; cc < CDIM; cc += 32) {
    __syncthreads();
    // Stage 32c x 64n tile of normalized activations into LDS (f16, transposed)
#pragma unroll
    for (int i = 0; i < 16; i++) {
      int flat = t * 16 + i;               // 0..2047
      int cl = flat >> 6;                  // 0..31
      int nl = flat & 63;                  // 0..63
      int c = cc + cl;
      int g = c >> 3;
      float mu = stats[(b * NGROUP + g) * 2 + 0];
      float rstd = stats[(b * NGROUP + g) * 2 + 1];
      float val = x[((size_t)b * CDIM + c) * NPOS + nbase + nl];
      val = (val - mu) * rstd * gamma[c] + beta[c];
      hs[nl][cl] = (_Float16)val;
    }
    __syncthreads();

    // A operand: 16 rows of W (o-major), K chunk of 32
    AF a;
#pragma unroll
    for (int v = 0; v < 8; v++) {
      int kb = 2 * v + 16 * half;
      a.u[v] = *(const unsigned int*)(w16 + (size_t)(obase + wv * 16 + col) * CDIM + cc + kb);
    }
#pragma unroll
    for (int nt = 0; nt < 4; nt++) {
      AF bop;
#pragma unroll
      for (int v = 0; v < 8; v++) {
        int kb = 2 * v + 16 * half;
        bop.u[v] = *(const unsigned int*)(&hs[nt * 16 + col][kb]);
      }
      acc[nt] = __builtin_amdgcn_wmma_f32_16x16x32_f16(
          false, a.v, false, bop.v, (short)0, acc[nt], false, false);
    }
  }

  // Bias + scatter: q,k -> (B,N,C); v -> (B,C,N)
#pragma unroll
  for (int nt = 0; nt < 4; nt++) {
    int n = nbase + nt * 16 + col;
    size_t rowb = ((size_t)b * NPOS + n) * CDIM;
#pragma unroll
    for (int j = 0; j < 8; j++) {
      int o = obase + wv * 16 + j + 8 * half;
      _Float16 hval = (_Float16)(acc[nt][j] + bq[o]);
      if (o < CDIM)            qt[rowb + o] = hval;
      else if (o < 2 * CDIM)   kt[rowb + (o - CDIM)] = hval;
      else                     vt[((size_t)b * CDIM + (o - 2 * CDIM)) * NPOS + n] = hval;
    }
  }
}

// ---------------------------------------------------------------------------
// Kernel 3: flash attention. 1 block = 64 query rows (4 waves x 16 rows),
// key blocks of 32, async LDS-DMA double buffering for K/V tiles.
// ---------------------------------------------------------------------------
__global__ __launch_bounds__(128) void attn(
    const _Float16* __restrict__ qt, const _Float16* __restrict__ kt,
    const _Float16* __restrict__ vt, _Float16* __restrict__ ot) {
  const int b = blockIdx.y;
  const int qbase = blockIdx.x * 64;
  const int t = threadIdx.x, wv = t >> 5, ln = t & 31;
  const int col = ln & 15, half = ln >> 4;

  __shared__ _Float16 qs[64 * QS_STR];        // ~33 KB   [qrow][c]
  __shared__ _Float16 ks[2][32 * KS_STR];     // ~33 KB   [key][c]
  __shared__ _Float16 vs[2][CDIM * VS_STR];   // ~40 KB   [c][key]
  __shared__ _Float16 ps[4][16][PS_STR];      //  5 KB    per-wave P scratch

  const unsigned qsA  = lds_off(&qs[0]);
  const unsigned ksA0 = lds_off(&ks[0][0]), ksA1 = lds_off(&ks[1][0]);
  const unsigned vsA0 = lds_off(&vs[0][0]), vsA1 = lds_off(&vs[1][0]);

  // --- issue async DMA: Q tile (64 rows x 512B) -> qs
#pragma unroll
  for (int i = 0; i < 16; i++) {
    int fl = t * 16 + i;                 // 0..2047
    int row = fl >> 5, q = fl & 31;
    async_b128(qsA + (unsigned)(row * QS_STR + q * 8) * 2,
               qt + ((size_t)b * NPOS + qbase + row) * CDIM + q * 8);
  }
  // --- issue async DMA: K/V tile 0
#pragma unroll
  for (int i = 0; i < 8; i++) {
    int fl = t * 8 + i;                  // 0..1023
    int row = fl >> 5, q = fl & 31;      // K: 32 rows x 32 xfers
    async_b128(ksA0 + (unsigned)(row * KS_STR + q * 8) * 2,
               kt + ((size_t)b * NPOS + row) * CDIM + q * 8);
    int c = fl >> 2, qq = fl & 3;        // V: 256 rows x 4 xfers
    async_b128(vsA0 + (unsigned)(c * VS_STR + qq * 8) * 2,
               vt + ((size_t)b * CDIM + c) * NPOS + qq * 8);
  }
  wait_async16();                        // first 16 (Q tile) complete, K/V0 in flight
  __syncthreads();

  // Preload Q A-fragments: 8 K-chunks of 32
  AF aq[8];
#pragma unroll
  for (int c8 = 0; c8 < 8; c8++) {
#pragma unroll
    for (int v = 0; v < 8; v++) {
      int kb = 2 * v + 16 * half;
      aq[c8].u[v] = *(const unsigned int*)(qs + (wv * 16 + col) * QS_STR + c8 * 32 + kb);
    }
  }

  v8f zero = {};
  v8f acc[16];
#pragma unroll
  for (int i = 0; i < 16; i++) acc[i] = zero;
  float mrow[8], lrow[8];
#pragma unroll
  for (int j = 0; j < 8; j++) { mrow[j] = -1e30f; lrow[j] = 0.0f; }

  for (int kb0 = 0; kb0 < NPOS; kb0 += 32) {
    const int cur = (kb0 >> 5) & 1;
    wait_async0();                       // this wave's K/V tile DMAs complete
    __syncthreads();                     // whole tile valid; prev buffer reads done

    // Prefetch next K/V tile into the other buffer (overlaps with compute)
    if (kb0 + 32 < NPOS) {
      const unsigned kn = cur ? ksA0 : ksA1;
      const unsigned vn = cur ? vsA0 : vsA1;
#pragma unroll
      for (int i = 0; i < 8; i++) {
        int fl = t * 8 + i;
        int row = fl >> 5, q = fl & 31;
        async_b128(kn + (unsigned)(row * KS_STR + q * 8) * 2,
                   kt + ((size_t)b * NPOS + kb0 + 32 + row) * CDIM + q * 8);
        int c = fl >> 2, qq = fl & 3;
        async_b128(vn + (unsigned)(c * VS_STR + qq * 8) * 2,
                   vt + ((size_t)b * CDIM + c) * NPOS + kb0 + 32 + qq * 8);
      }
    }

    const _Float16* ksc = &ks[cur][0];
    const _Float16* vsc = &vs[cur][0];

    // S = Q * K^T  (two 16x16 tiles over 32 keys)
    v8f s0 = zero, s1 = zero;
#pragma unroll
    for (int c8 = 0; c8 < 8; c8++) {
      AF b0, b1;
#pragma unroll
      for (int v = 0; v < 8; v++) {
        int kbase = 2 * v + 16 * half;
        b0.u[v] = *(const unsigned int*)(ksc + (size_t)col * KS_STR + c8 * 32 + kbase);
        b1.u[v] = *(const unsigned int*)(ksc + (size_t)(16 + col) * KS_STR + c8 * 32 + kbase);
      }
      s0 = __builtin_amdgcn_wmma_f32_16x16x32_f16(false, aq[c8].v, false, b0.v, (short)0, s0, false, false);
      s1 = __builtin_amdgcn_wmma_f32_16x16x32_f16(false, aq[c8].v, false, b1.v, (short)0, s1, false, false);
    }

    // Online softmax (rows in wave halves; cols = lanes within half)
    const float scale = 0.0625f;   // 1/sqrt(256)
    float alpha[8];
#pragma unroll
    for (int j = 0; j < 8; j++) {
      float a0 = s0[j] * scale, a1 = s1[j] * scale;
      float rmax = red16_max(fmaxf(a0, a1));
      float mn = fmaxf(mrow[j], rmax);
      float al = __expf(mrow[j] - mn);
      float p0 = __expf(a0 - mn);
      float p1 = __expf(a1 - mn);
      ps[wv][j + 8 * half][col]      = (_Float16)p0;
      ps[wv][j + 8 * half][16 + col] = (_Float16)p1;
      float rsum = red16_sum(p0 + p1);
      lrow[j] = lrow[j] * al + rsum;
      mrow[j] = mn;
      alpha[j] = al;
    }
    // per-wave LDS scratch: prevent compiler reordering of the A-frag reads
    __asm__ volatile("" ::: "memory");
    __builtin_amdgcn_wave_barrier();

    // P as A operand (16 x 32)
    AF ap;
#pragma unroll
    for (int v = 0; v < 8; v++) {
      int kbA = 2 * v + 16 * half;
      ap.u[v] = *(const unsigned int*)(&ps[wv][col][kbA]);
    }

    // O = O*alpha + P * V  (16 column tiles of 16), V tile is [c][key]
#pragma unroll
    for (int ct = 0; ct < 16; ct++) {
      AF bv;
#pragma unroll
      for (int v = 0; v < 8; v++) {
        int kk = 2 * v + 16 * half;
        bv.u[v] = *(const unsigned int*)(vsc + (size_t)(ct * 16 + col) * VS_STR + kk);
      }
      v8f cin;
#pragma unroll
      for (int j = 0; j < 8; j++) cin[j] = acc[ct][j] * alpha[j];
      acc[ct] = __builtin_amdgcn_wmma_f32_16x16x32_f16(
          false, ap.v, false, bv.v, (short)0, cin, false, false);
    }
  }

  // Normalize and write O in (B,N,C) f16
#pragma unroll
  for (int ct = 0; ct < 16; ct++) {
#pragma unroll
    for (int j = 0; j < 8; j++) {
      int r = qbase + wv * 16 + j + 8 * half;
      float o = acc[ct][j] / lrow[j];
      ot[((size_t)b * NPOS + r) * CDIM + ct * 16 + col] = (_Float16)o;
    }
  }
}

// ---------------------------------------------------------------------------
// Kernel 4: projection GEMM + bias + residual (f32 output)
// ---------------------------------------------------------------------------
__global__ __launch_bounds__(128) void proj_gemm(
    const float* __restrict__ x, const _Float16* __restrict__ ot,
    const _Float16* __restrict__ wp16, const float* __restrict__ bp,
    float* __restrict__ out) {
  const int b = blockIdx.z;
  const int obase = blockIdx.y * 64;
  const int nbase = blockIdx.x * 64;
  const int t = threadIdx.x, wv = t >> 5, ln = t & 31;
  const int col = ln & 15, half = ln >> 4;

  __shared__ _Float16 os[64][OS_STR];   // [n_local][c_local], 80B row stride
  const unsigned osA = lds_off(&os[0][0]);

  v8f zero = {};
  v8f acc[4];
#pragma unroll
  for (int i = 0; i < 4; i++) acc[i] = zero;

  for (int cc = 0; cc < CDIM; cc += 32) {
    __syncthreads();
    // Async stage 64n x 32c tile of O ((N,C) layout): 256 b128 DMAs
#pragma unroll
    for (int i = 0; i < 2; i++) {
      int fl = t * 2 + i;                  // 0..255
      int row = fl >> 2, q = fl & 3;
      async_b128(osA + (unsigned)(row * OS_STR + q * 8) * 2,
                 ot + ((size_t)b * NPOS + nbase + row) * CDIM + cc + q * 8);
    }
    wait_async0();
    __syncthreads();

    AF a;
#pragma unroll
    for (int v = 0; v < 8; v++) {
      int kb = 2 * v + 16 * half;
      a.u[v] = *(const unsigned int*)(wp16 + (size_t)(obase + wv * 16 + col) * CDIM + cc + kb);
    }
#pragma unroll
    for (int nt = 0; nt < 4; nt++) {
      AF bop;
#pragma unroll
      for (int v = 0; v < 8; v++) {
        int kb = 2 * v + 16 * half;
        bop.u[v] = *(const unsigned int*)(&os[nt * 16 + col][kb]);
      }
      acc[nt] = __builtin_amdgcn_wmma_f32_16x16x32_f16(
          false, a.v, false, bop.v, (short)0, acc[nt], false, false);
    }
  }

#pragma unroll
  for (int nt = 0; nt < 4; nt++) {
    int n = nbase + nt * 16 + col;
#pragma unroll
    for (int j = 0; j < 8; j++) {
      int o = obase + wv * 16 + j + 8 * half;
      size_t gi = ((size_t)b * CDIM + o) * NPOS + n;
      out[gi] = x[gi] + acc[nt][j] + bp[o];
    }
  }
}

// ---------------------------------------------------------------------------
extern "C" void kernel_launch(void* const* d_in, const int* in_sizes, int n_in,
                              void* d_out, int out_size, void* d_ws, size_t ws_size,
                              hipStream_t stream) {
  const float* x     = (const float*)d_in[0];
  const float* gamma = (const float*)d_in[1];
  const float* beta  = (const float*)d_in[2];
  const float* wqkv  = (const float*)d_in[3];
  const float* bqkv  = (const float*)d_in[4];
  const float* wproj = (const float*)d_in[5];
  const float* bproj = (const float*)d_in[6];
  float* out = (float*)d_out;

  // Workspace layout (needs ~36 MB)
  char* ws = (char*)d_ws;
  float*    stats = (float*)ws;                               // 256 f32
  _Float16* wq16  = (_Float16*)(ws + ((size_t)1  << 20));     // 384 KB
  _Float16* wp16  = (_Float16*)(ws + ((size_t)2  << 20));     // 128 KB
  _Float16* qt    = (_Float16*)(ws + ((size_t)4  << 20));     // 8 MB (B,N,C)
  _Float16* kt    = (_Float16*)(ws + ((size_t)12 << 20));     // 8 MB (B,N,C)
  _Float16* vt    = (_Float16*)(ws + ((size_t)20 << 20));     // 8 MB (B,C,N)
  _Float16* ot    = (_Float16*)(ws + ((size_t)28 << 20));     // 8 MB (B,N,C)

  cvt_weights<<<dim3(768), dim3(256), 0, stream>>>(wqkv, wproj, wq16, wp16);
  gn_stats<<<dim3(BATCH * NGROUP), dim3(256), 0, stream>>>(x, stats);
  qkv_gemm<<<dim3(NPOS / 64, (3 * CDIM) / 64, BATCH), dim3(128), 0, stream>>>(
      x, gamma, beta, stats, wq16, bqkv, qt, kt, vt);
  attn<<<dim3(NPOS / 64, BATCH), dim3(128), 0, stream>>>(qt, kt, vt, ot);
  proj_gemm<<<dim3(NPOS / 64, CDIM / 64, BATCH), dim3(128), 0, stream>>>(
      x, ot, wp16, bproj, out);
}
